// LANLayer_60945585931037
// MI455X (gfx1250) — matrix operand order repaired and checked by reading, not compiled, not run on hardware
//
#include <hip/hip_runtime.h>
#include <hip/hip_fp16.h>
#include <stdint.h>

// ---------------------------------------------------------------------------
// LAN layer for MI455X (gfx1250, wave32, WMMA).
//
// Pipeline (FP16-in / FP32-accumulate WMMA; softmax eliminated by
// monotonicity — top-k of softmax(sim/T) == top-k of the raw dot product):
//   K0  convert x -> f16, transpose W_proj/W_conv -> f16 (for B-fragments)
//   K1  h = relu(x @ W_proj + b); row norms; emit h_f16 and hn_f16
//   K2  fused  hn @ hn^T  +  per-row running top-10  (no 256MB sim matrix)
//       32 fixed rows per WG -> each swept A-tile feeds 2 accumulators,
//       halving L2 traffic (0.5 GB total) vs a 16-row blocking.
//   K3  hw = h @ W_conv
//   K4  out = relu(D^-1/2 (A+I) D^-1/2 @ hw)   -- <=11 nnz/row gather-SpMM
// ---------------------------------------------------------------------------

typedef __attribute__((ext_vector_type(16))) _Float16     v16h;
typedef __attribute__((ext_vector_type(8)))  float        v8f;
typedef __attribute__((ext_vector_type(4)))  unsigned int v4u;

#define N_PTS 8192
#define DIM   256
#define HID   128
#define KSEL  10

union AFrag { v16h h; unsigned int u[8]; };
union BFrag { v16h h; v4u q[2]; };

// 16-bit A fragment (16x32, M x K): lane m = lane&15, hi = lane>>4.
// VGPR i: K = 2i + 16*hi (pairs; contiguous, matches load_b_frag).
// (Compiler coalesces these into two b128 loads per fragment.)
__device__ __forceinline__ v16h load_a_frag(const _Float16* __restrict__ base,
                                            int pitch, int m, int hi, int k0) {
  AFrag a;
  const _Float16* p = base + m * pitch + k0;
#pragma unroll
  for (int i = 0; i < 8; ++i) {
    const int kb = 2 * i + 16 * hi;
    a.u[i] = *(const unsigned int*)(p + kb);
  }
  return a.h;
}

// 16-bit B fragment (32x16, K x N) from a TRANSPOSED (N-major) array:
// lane n = lane&15, hi = lane>>4; halves j=0..15 hold K = k0 + 16*hi + j,
// which is contiguous in the transposed storage -> two b128 loads.
__device__ __forceinline__ v16h load_b_frag(const _Float16* __restrict__ baseT,
                                            int pitch, int n, int hi, int k0) {
  BFrag b;
  const _Float16* p = baseT + n * pitch + k0 + 16 * hi;
  b.q[0] = *(const v4u*)(p);
  b.q[1] = *(const v4u*)(p + 8);
  return b.h;
}

// Register-resident sorted top-10 (descending). Guard compare is false for
// almost every candidate; bubble chain is fully unrolled (constant indices).
struct TopK {
  float v[KSEL];
  int   id[KSEL];
  __device__ __forceinline__ void init() {
#pragma unroll
    for (int s = 0; s < KSEL; ++s) { v[s] = -3.0e38f; id[s] = -1; }
  }
  __device__ __forceinline__ void insert(float x, int xi) {
    if (x > v[KSEL - 1]) {
      v[KSEL - 1] = x; id[KSEL - 1] = xi;
#pragma unroll
      for (int p = KSEL - 1; p >= 1; --p) {
        if (v[p] > v[p - 1]) {
          float tv = v[p]; v[p] = v[p - 1]; v[p - 1] = tv;
          int   ti = id[p]; id[p] = id[p - 1]; id[p - 1] = ti;
        }
      }
    }
  }
};

// ---------------------------------------------------------------- K0 helpers
__global__ void convert_x_kernel(const float* __restrict__ x,
                                 _Float16* __restrict__ x16, int n) {
  for (int i = blockIdx.x * blockDim.x + threadIdx.x; i < n;
       i += gridDim.x * blockDim.x)
    x16[i] = (_Float16)x[i];
}

// w: [rows][cols] f32 -> wt: [cols][rows] f16
__global__ void transpose_w_kernel(const float* __restrict__ w,
                                   _Float16* __restrict__ wt, int rows, int cols) {
  const int n = rows * cols;
  for (int i = blockIdx.x * blockDim.x + threadIdx.x; i < n;
       i += gridDim.x * blockDim.x) {
    const int r = i / cols, c = i % cols;
    wt[c * rows + r] = (_Float16)w[i];
  }
}

// -------------------------------------------------- K1: proj + relu + norms
// Block = 256 threads (8 waves), 16 rows x 128 cols; wave w owns col tile w.
__global__ __launch_bounds__(256) void proj_kernel(
    const _Float16* __restrict__ x16, const _Float16* __restrict__ wtp,
    const float* __restrict__ bias, _Float16* __restrict__ h16,
    _Float16* __restrict__ hn16) {
  __shared__ float hbuf[16][HID + 1];  // +1 pad: conflict-free column sweep
  __shared__ float ninvs[16];
  const int row0 = blockIdx.x * 16;
  const int tid  = threadIdx.x;
  const int wave = tid >> 5, lane = tid & 31;
  const int nn = lane & 15, hi = lane >> 4;
  const int cw = wave * 16;

  v8f c = {};
#pragma unroll
  for (int kk = 0; kk < DIM / 32; ++kk) {
    v16h a = load_a_frag(x16 + row0 * DIM, DIM, nn, hi, kk * 32);
    v16h b = load_b_frag(wtp + cw * DIM, DIM, nn, hi, kk * 32);
    c = __builtin_amdgcn_wmma_f32_16x16x32_f16(false, a, false, b,
                                               (short)0, c, false, false);
  }
  const int   col = cw + nn;
  const float bv  = bias[col];
#pragma unroll
  for (int r = 0; r < 8; ++r) {
    float v = c[r] + bv;
    v = v > 0.f ? v : 0.f;                 // relu
    hbuf[r + 8 * hi][col] = v;
  }
  __syncthreads();
  if (tid < 16) {
    float s = 0.f;
    for (int cc = 0; cc < HID; ++cc) { float v = hbuf[tid][cc]; s += v * v; }
    ninvs[tid] = 1.0f / fmaxf(sqrtf(s), 1e-12f);
  }
  __syncthreads();
  for (int e = tid; e < 16 * HID; e += 256) {
    const int rr = e >> 7, cc = e & (HID - 1);
    const float v = hbuf[rr][cc];
    h16[(row0 + rr) * HID + cc]  = (_Float16)v;
    hn16[(row0 + rr) * HID + cc] = (_Float16)(v * ninvs[rr]);
  }
}

// ------------------------------------- K2: fused hn @ hn^T + per-row top-10
// 32 fixed rows sit on the WMMA *N* axis (two 16-row B blocks) so each
// lane's accumulator values belong to exactly one row -> private register
// top-10 lists, no transpose. Each swept A-tile (16 cols x 128 K) feeds two
// WMMA chains -> 2x arithmetic intensity, 0.5 GB total L2 traffic.
// 8 waves sweep disjoint 1024-column stripes; 16 partial lists per row are
// merged through LDS at the end. hn_f16 (2 MB) is L2-resident.
__global__ __launch_bounds__(256) void sim_topk_kernel(
    const _Float16* __restrict__ hn16, int* __restrict__ topi,
    float* __restrict__ dinv) {
  __shared__ float lv[32][16 * KSEL];
  __shared__ int   li[32][16 * KSEL];
  const int row0 = blockIdx.x * 32;
  const int tid  = threadIdx.x;
  const int wave = tid >> 5, lane = tid & 31;
  const int nn = lane & 15, hi = lane >> 4;

  // Fixed-row operands (B): loaded once, live in 64 VGPRs.
  v16h bfrag0[4], bfrag1[4];
#pragma unroll
  for (int kk = 0; kk < 4; ++kk) {
    bfrag0[kk] = load_b_frag(hn16 + row0 * HID, HID, nn, hi, kk * 32);
    bfrag1[kk] = load_b_frag(hn16 + (row0 + 16) * HID, HID, nn, hi, kk * 32);
  }

  TopK tk0, tk1;
  tk0.init();
  tk1.init();

  const int tiles_per_wave = (N_PTS / 16) / 8;  // 64
  const int t0 = wave * tiles_per_wave;
  for (int t = 0; t < tiles_per_wave; ++t) {
    const int colbase = (t0 + t) * 16;
    // Prefetch next column tile's row for this lane (global_prefetch_b8).
    if (t + 1 < tiles_per_wave)
      __builtin_prefetch(hn16 + (colbase + 16 + nn) * HID, 0, 3);

    v16h a[4];
#pragma unroll
    for (int kk = 0; kk < 4; ++kk)
      a[kk] = load_a_frag(hn16 + colbase * HID, HID, nn, hi, kk * 32);

    v8f c0 = {}, c1 = {};
#pragma unroll
    for (int kk = 0; kk < 4; ++kk) {
      c0 = __builtin_amdgcn_wmma_f32_16x16x32_f16(false, a[kk], false,
                                                  bfrag0[kk], (short)0, c0,
                                                  false, false);
      c1 = __builtin_amdgcn_wmma_f32_16x16x32_f16(false, a[kk], false,
                                                  bfrag1[kk], (short)0, c1,
                                                  false, false);
    }
    // C layout: N = lane&15 (fixed row), M = r + 8*(lane>>4) (sweep column).
#pragma unroll
    for (int r = 0; r < 8; ++r) {
      const int colidx = colbase + r + 8 * hi;
      tk0.insert(c0[r], colidx);
      tk1.insert(c1[r], colidx);
    }
  }

  // Dump 16 partial lists per row (8 waves x 2 half-lanes) and merge.
  const int p = wave * 2 + hi;
#pragma unroll
  for (int s = 0; s < KSEL; ++s) {
    lv[nn][p * KSEL + s]      = tk0.v[s];
    li[nn][p * KSEL + s]      = tk0.id[s];
    lv[nn + 16][p * KSEL + s] = tk1.v[s];
    li[nn + 16][p * KSEL + s] = tk1.id[s];
  }
  __syncthreads();
  if (tid < 32) {
    TopK f; f.init();
    for (int s = 0; s < 16 * KSEL; ++s) f.insert(lv[tid][s], li[tid][s]);
    const int rg  = row0 + tid;
    int       deg = 1;  // the +I diagonal
#pragma unroll
    for (int s = 0; s < KSEL; ++s) {
      topi[rg * KSEL + s] = f.id[s];
      if (f.id[s] != rg) deg++;  // adj diagonal is zeroed in the reference
    }
    dinv[rg] = rsqrtf((float)deg);
  }
}

// ----------------------------------------------------- K3: hw = h @ W_conv
__global__ __launch_bounds__(256) void conv_gemm_kernel(
    const _Float16* __restrict__ h16, const _Float16* __restrict__ wtc,
    float* __restrict__ hw) {
  const int row0 = blockIdx.x * 16;
  const int tid  = threadIdx.x;
  const int wave = tid >> 5, lane = tid & 31;
  const int nn = lane & 15, hi = lane >> 4;
  const int cw = wave * 16;
  v8f c = {};
#pragma unroll
  for (int kk = 0; kk < HID / 32; ++kk) {
    v16h a = load_a_frag(h16 + row0 * HID, HID, nn, hi, kk * 32);
    v16h b = load_b_frag(wtc + cw * HID, HID, nn, hi, kk * 32);
    c = __builtin_amdgcn_wmma_f32_16x16x32_f16(false, a, false, b,
                                               (short)0, c, false, false);
  }
#pragma unroll
  for (int r = 0; r < 8; ++r)
    hw[(row0 + r + 8 * hi) * HID + cw + nn] = c[r];
}

// --------------------- K4: out = relu(D^-1/2 (A+I) D^-1/2 @ hw), <=11 nnz/row
__global__ __launch_bounds__(128) void gcn_gather_kernel(
    const float* __restrict__ hw, const int* __restrict__ topi,
    const float* __restrict__ dinv, float* __restrict__ out) {
  const int   i  = blockIdx.x;
  const int   t  = threadIdx.x;
  const float di = dinv[i];
  float acc = di * di * hw[i * HID + t];  // A_hat[i,i] = 1
#pragma unroll
  for (int s = 0; s < KSEL; ++s) {
    const int j = topi[i * KSEL + s];
    if (j != i) acc += di * dinv[j] * hw[j * HID + t];
  }
  out[i * HID + t] = acc > 0.f ? acc : 0.f;
}

// ---------------------------------------------------------------------------
extern "C" void kernel_launch(void* const* d_in, const int* in_sizes, int n_in,
                              void* d_out, int out_size, void* d_ws,
                              size_t ws_size, hipStream_t stream) {
  const float* x  = (const float*)d_in[0];  // [8192,256]
  const float* Wp = (const float*)d_in[1];  // [256,128]
  const float* bp = (const float*)d_in[2];  // [128]
  const float* Wc = (const float*)d_in[3];  // [128,128]
  float*       out = (float*)d_out;         // [8192,128]

  // Workspace carve-up (~12.7 MB total, 256B-aligned slabs).
  char*  ws  = (char*)d_ws;
  size_t off = 0;
  auto alloc = [&](size_t bytes) -> void* {
    void* p = ws + off;
    off += (bytes + 255) & ~(size_t)255;
    return p;
  };
  _Float16* x16  = (_Float16*)alloc((size_t)N_PTS * DIM * 2);
  _Float16* wtp  = (_Float16*)alloc((size_t)HID * DIM * 2);
  _Float16* wtc  = (_Float16*)alloc((size_t)HID * HID * 2);
  _Float16* h16  = (_Float16*)alloc((size_t)N_PTS * HID * 2);
  _Float16* hn16 = (_Float16*)alloc((size_t)N_PTS * HID * 2);
  float*    hw   = (float*)alloc((size_t)N_PTS * HID * 4);
  int*      topi = (int*)alloc((size_t)N_PTS * KSEL * 4);
  float*    dinv = (float*)alloc((size_t)N_PTS * 4);
  (void)ws_size; (void)in_sizes; (void)n_in; (void)out_size;

  convert_x_kernel<<<512, 256, 0, stream>>>(x, x16, N_PTS * DIM);
  transpose_w_kernel<<<128, 256, 0, stream>>>(Wp, wtp, DIM, HID);
  transpose_w_kernel<<<64, 256, 0, stream>>>(Wc, wtc, HID, HID);
  proj_kernel<<<N_PTS / 16, 256, 0, stream>>>(x16, wtp, bp, h16, hn16);
  sim_topk_kernel<<<N_PTS / 32, 256, 0, stream>>>(hn16, topi, dinv);
  conv_gemm_kernel<<<N_PTS / 16, 256, 0, stream>>>(h16, wtc, hw);
  gcn_gather_kernel<<<N_PTS, 128, 0, stream>>>(hw, topi, dinv, out);
}
